// SequenceMemoryUpdater_71365176590515
// MI455X (gfx1250) — matrix operand order repaired and checked by reading, not compiled, not run on hardware
//
#include <hip/hip_runtime.h>

typedef __attribute__((ext_vector_type(16))) __bf16       v16bf;
typedef __attribute__((ext_vector_type(8)))  float        v8f;
typedef __attribute__((ext_vector_type(8)))  unsigned int v8u;

static __device__ __forceinline__ unsigned short f2bf(float f) {
  unsigned int u = __float_as_uint(f);
  u += 0x7FFFu + ((u >> 16) & 1u);          // round-to-nearest-even
  return (unsigned short)(u >> 16);
}
static __device__ __forceinline__ unsigned int pack2(float lo, float hi) {
  return (unsigned int)f2bf(lo) | ((unsigned int)f2bf(hi) << 16);
}
static __device__ __forceinline__ v8f wmma_bf16(v8u a, v8u b, v8f c) {
  return __builtin_amdgcn_wmma_f32_16x16x32_bf16(
      false, __builtin_bit_cast(v16bf, a),
      false, __builtin_bit_cast(v16bf, b),
      (short)0, c, false, false);
}

// ---- weight conversion: f32 -> bf16 in workspace --------------------------
// ws layout (ushort): [0, 256*256)          W_cat[:, :256]
//                     [WCAT, WCAT+768*256)  W_ih
//                     [WCAT+WIH, ...)       W_hh
#define WCAT_E (256 * 256)
#define WIH_E  (768 * 256)

__global__ void convert_weights_kernel(const float* __restrict__ Wcat,
                                       const float* __restrict__ Wih,
                                       const float* __restrict__ Whh,
                                       unsigned short* __restrict__ ws) {
  int i = blockIdx.x * blockDim.x + threadIdx.x;
  if (i < WCAT_E) {
    int r = i >> 8, c = i & 255;
    ws[i] = f2bf(Wcat[r * 257 + c]);
  } else if (i < WCAT_E + WIH_E) {
    int j = i - WCAT_E;
    ws[WCAT_E + j] = f2bf(Wih[j]);
  } else if (i < WCAT_E + 2 * WIH_E) {
    int j = i - WCAT_E - WIH_E;
    ws[WCAT_E + WIH_E + j] = f2bf(Whh[j]);
  }
}

// ---- fused cat-projection + GRU cell + scatter, one wave per 16 rows ------
__global__ void __launch_bounds__(32)
gru_update_kernel(const int*   __restrict__ ids,
                  const float* __restrict__ msg,
                  const float* __restrict__ ts,
                  const float* __restrict__ amt,
                  const float* __restrict__ mem,
                  const float* __restrict__ Wcat_f32,
                  const float* __restrict__ bcat,
                  const float* __restrict__ bih,
                  const float* __restrict__ bhh,
                  const unsigned short* __restrict__ ws,
                  float* __restrict__ out_mem,
                  float* __restrict__ out_lu,
                  int n_upd) {
  __shared__ unsigned short lds_msg[16 * 256];  // bf16 message tile
  __shared__ unsigned short lds_h[16 * 256];    // bf16 gathered h tile
  __shared__ unsigned short lds_x[16 * 256];    // bf16 projected x tile

  const int lane = threadIdx.x & 31;
  const int lrow = lane & 15;
  const int lhi  = lane >> 4;
  const int base = blockIdx.x * 16;

  const unsigned short* wsWcat = ws;
  const unsigned short* wsWih  = ws + WCAT_E;
  const unsigned short* wsWhh  = ws + WCAT_E + WIH_E;

  // per-lane C-fragment row metadata: vgpr i <-> row m = lhi*8 + i
  int   idv[8];
  float amtv[8];
#pragma unroll
  for (int i = 0; i < 8; ++i) {
    int gm  = base + lhi * 8 + i;
    int gmc = gm < n_upd ? gm : n_upd - 1;
    idv[i]  = ids[gmc];
    amtv[i] = amt[gmc];
  }

  // stage message + gathered h tiles to LDS as bf16 (coalesced b128 loads)
  for (int t = lane; t < 16 * 64; t += 32) {
    int row = t >> 6;
    int c   = (t & 63) * 4;
    int gm  = base + row;
    int gmc = gm < n_upd ? gm : n_upd - 1;
    float4 mv = *(const float4*)(msg + (size_t)gmc * 256 + c);
    int   nid = ids[gmc];
    float4 hv = *(const float4*)(mem + (size_t)nid * 256 + c);
    unsigned int* pm = (unsigned int*)&lds_msg[row * 256 + c];
    pm[0] = pack2(mv.x, mv.y);
    pm[1] = pack2(mv.z, mv.w);
    unsigned int* ph = (unsigned int*)&lds_h[row * 256 + c];
    ph[0] = pack2(hv.x, hv.y);
    ph[1] = pack2(hv.z, hv.w);
  }
  __syncthreads();

  // ---- stage 1: x = cat(msg, amt) @ W_cat.T + b_cat  (K=256 WMMA + rank-1)
#pragma unroll 1
  for (int nt = 0; nt < 16; ++nt) {
    int   n   = nt * 16 + lrow;
    float bc  = bcat[n];
    float wcl = Wcat_f32[n * 257 + 256];     // the amount column
    v8f c;
#pragma unroll
    for (int i = 0; i < 8; ++i) c[i] = bc + amtv[i] * wcl;

#pragma unroll 1
    for (int kt = 0; kt < 8; ++kt) {
      v8u a, b;
#pragma unroll
      for (int j = 0; j < 8; ++j) {
        int ka = kt * 32 + lhi * 8 + ((j < 4) ? 2 * j : 16 + 2 * (j - 4));
        a[j] = *(const unsigned int*)&lds_msg[lrow * 256 + ka];
        int kb = kt * 32 + lhi * 8 + ((j < 4) ? 2 * j : 16 + 2 * (j - 4));
        b[j] = *(const unsigned int*)&wsWcat[n * 256 + kb];
      }
      c = wmma_bf16(a, b, c);
    }
#pragma unroll
    for (int i = 0; i < 8; ++i)
      lds_x[(lhi * 8 + i) * 256 + n] = f2bf(c[i]);
  }
  __syncthreads();

  // ---- stage 2: GRU gates (6 WMMAs per k-tile) + elementwise + scatter ----
#pragma unroll 1
  for (int nt = 0; nt < 16; ++nt) {
    int   n   = nt * 16 + lrow;
    float br  = bih[n]       + bhh[n];
    float bz  = bih[256 + n] + bhh[256 + n];
    float bi  = bih[512 + n];
    float bh  = bhh[512 + n];
    v8f cr, cz, cin, chn;
#pragma unroll
    for (int i = 0; i < 8; ++i) { cr[i] = br; cz[i] = bz; cin[i] = bi; chn[i] = bh; }

#pragma unroll 1
    for (int kt = 0; kt < 8; ++kt) {
      v8u ax, ah;
#pragma unroll
      for (int j = 0; j < 8; ++j) {
        int ka = kt * 32 + lhi * 8 + ((j < 4) ? 2 * j : 16 + 2 * (j - 4));
        ax[j] = *(const unsigned int*)&lds_x[lrow * 256 + ka];
        ah[j] = *(const unsigned int*)&lds_h[lrow * 256 + ka];
      }
      v8u wir, wiz, win, whr, whz, whn;
#pragma unroll
      for (int j = 0; j < 8; ++j) {
        int kb = kt * 32 + lhi * 8 + ((j < 4) ? 2 * j : 16 + 2 * (j - 4));
        wir[j] = *(const unsigned int*)&wsWih[(0   + n) * 256 + kb];
        wiz[j] = *(const unsigned int*)&wsWih[(256 + n) * 256 + kb];
        win[j] = *(const unsigned int*)&wsWih[(512 + n) * 256 + kb];
        whr[j] = *(const unsigned int*)&wsWhh[(0   + n) * 256 + kb];
        whz[j] = *(const unsigned int*)&wsWhh[(256 + n) * 256 + kb];
        whn[j] = *(const unsigned int*)&wsWhh[(512 + n) * 256 + kb];
      }
      cr  = wmma_bf16(ax, wir, cr);
      cr  = wmma_bf16(ah, whr, cr);
      cz  = wmma_bf16(ax, wiz, cz);
      cz  = wmma_bf16(ah, whz, cz);
      cin = wmma_bf16(ax, win, cin);
      chn = wmma_bf16(ah, whn, chn);
    }

    // gates in f32, blend against f32 h from global, scatter to output
#pragma unroll
    for (int i = 0; i < 8; ++i) {
      int gm = base + lhi * 8 + i;
      if (gm < n_upd) {
        float r  = 1.0f / (1.0f + __expf(-cr[i]));
        float z  = 1.0f / (1.0f + __expf(-cz[i]));
        float nn = tanhf(cin[i] + r * chn[i]);
        float hv = mem[(size_t)idv[i] * 256 + n];
        out_mem[(size_t)idv[i] * 256 + n] = (1.0f - z) * nn + z * hv;
      }
    }
  }

  // last_update scatter (one row per lane, lanes 0..15)
  if (lane < 16) {
    int gm = base + lane;
    if (gm < n_upd) out_lu[ids[gm]] = ts[gm];
  }
}

extern "C" void kernel_launch(void* const* d_in, const int* in_sizes, int n_in,
                              void* d_out, int out_size, void* d_ws, size_t ws_size,
                              hipStream_t stream) {
  const int*   ids  = (const int*)d_in[0];
  const float* msg  = (const float*)d_in[1];
  const float* ts   = (const float*)d_in[2];
  const float* amt  = (const float*)d_in[3];
  const float* mem  = (const float*)d_in[4];
  const float* lu   = (const float*)d_in[5];
  const float* Wcat = (const float*)d_in[6];
  const float* bcat = (const float*)d_in[7];
  const float* Wih  = (const float*)d_in[8];
  const float* Whh  = (const float*)d_in[9];
  const float* bih  = (const float*)d_in[10];
  const float* bhh  = (const float*)d_in[11];

  const int n_upd   = in_sizes[0];
  const int n_nodes = in_sizes[5];

  float* out_mem = (float*)d_out;
  float* out_lu  = out_mem + (size_t)n_nodes * 256;

  // bulk passthrough copies (overwritten rows scattered afterwards, same stream)
  hipMemcpyAsync(out_mem, mem, (size_t)n_nodes * 256 * sizeof(float),
                 hipMemcpyDeviceToDevice, stream);
  hipMemcpyAsync(out_lu, lu, (size_t)n_nodes * sizeof(float),
                 hipMemcpyDeviceToDevice, stream);

  unsigned short* ws = (unsigned short*)d_ws;
  {
    int total  = WCAT_E + 2 * WIH_E;
    int blocks = (total + 255) / 256;
    convert_weights_kernel<<<blocks, 256, 0, stream>>>(Wcat, Wih, Whh, ws);
  }
  {
    int tiles = (n_upd + 15) / 16;
    gru_update_kernel<<<tiles, 32, 0, stream>>>(
        ids, msg, ts, amt, mem, Wcat, bcat, bih, bhh, ws,
        out_mem, out_lu, n_upd);
  }
}
